// Spiking_Self_Attention_36670430773401
// MI455X (gfx1250) — matrix operand order repaired and checked by reading, not compiled, not run on hardware
//
// Spiking Self-Attention forward for MI455X (gfx1250, wave32, WMMA f16->f32).
// Pipeline: LIF -> 3x [WMMA GEMM -> BN stats -> BN apply(/LIF)] -> kv WMMA ->
//           attn WMMA -> LIF(0.5) -> proj WMMA GEMM -> BN stats -> BN+LIF -> out.
// All WMMA fragments load contiguously from global thanks to (tb,n,c) /
// (tb,c,n) layout choices; no LDS transposes required.
#include <hip/hip_runtime.h>

typedef __attribute__((ext_vector_type(16))) _Float16 v16h;
typedef __attribute__((ext_vector_type(8)))  _Float16 v8h;
typedef __attribute__((ext_vector_type(8)))  float    v8f;
typedef __attribute__((ext_vector_type(4)))  float    v4f;

#define TT 4
#define BB 16
#define CC 512
#define NN 1024
#define HH 8
#define DD 64
#define TB (TT*BB)

__device__ __forceinline__ v8f wmma_f16(v16h a, v16h b, v8f c) {
  // v_wmma_f32_16x16x32_f16: (neg_a, A, neg_b, B, c_mod, C, reuse_a, reuse_b)
  return __builtin_amdgcn_wmma_f32_16x16x32_f16(false, a, false, b, (short)0, c,
                                                false, false);
}

// A fragment (16x32 f16, M rows x K): lane L holds row (L&15);
// lanes 0-15: K = {0..7, 16..23}; lanes 16-31: K = {8..15, 24..31}.
// Source is row-major with leading dim `ld` halves (multiple of 8).
__device__ __forceinline__ v16h load_afrag(const _Float16* __restrict__ base,
                                           int ld, int lane) {
  const int row = lane & 15;
  const int kb  = (lane & 16) ? 8 : 0;
  const _Float16* p = base + row * ld + kb;
  v8h lo = *(const v8h*)(p);        // K = kb .. kb+7
  v8h hi = *(const v8h*)(p + 16);   // K = kb+16 .. kb+23
  v16h r;
#pragma unroll
  for (int j = 0; j < 8; ++j) { r[j] = lo[j]; r[8 + j] = hi[j]; }
  return r;
}

// B fragment (32x16 f16, K x N-cols): lane L holds col (L&15);
// lanes 0-15: K = {0..7, 16..23}; lanes 16-31: K = {8..15, 24..31}.
// Source stored column-rows: X[col][k] contiguous in k, leading dim `ld`.
__device__ __forceinline__ v16h load_bfrag(const _Float16* __restrict__ base,
                                           int ld, int lane) {
  const int col = lane & 15;
  const int kb  = (lane & 16) ? 8 : 0;
  const _Float16* p = base + col * ld + kb;
  v8h lo = *(const v8h*)(p);
  v8h hi = *(const v8h*)(p + 16);
  v16h r;
#pragma unroll
  for (int j = 0; j < 8; ++j) { r[j] = lo[j]; r[8 + j] = hi[j]; }
  return r;
}

// ---------------- elementwise / LIF kernels ----------------

__global__ void w2h_kernel(const float* __restrict__ w, _Float16* __restrict__ wh) {
  const int i = blockIdx.x * 256 + threadIdx.x;
  wh[i] = (_Float16)w[i];
}

// LIF over q and k. Reads (t,b,c,n) coalesced; writes spikes transposed to
// (tb, n, c) (GEMM-B layout) and final k membrane state (b,n,c).
__global__ void __launch_bounds__(256) lif_qk_kernel(
    const float* __restrict__ q, const float* __restrict__ k,
    _Float16* __restrict__ qs, _Float16* __restrict__ ks,
    float* __restrict__ vst) {
  const int i = blockIdx.x * 256 + threadIdx.x;  // (b,c,n), n fastest
  const int n = i & (NN - 1);
  const int c = (i >> 10) & (CC - 1);
  const int b = i >> 19;
  const int PLANE = BB * CC * NN;
  float vq = 0.f, vk = 0.f;
#pragma unroll
  for (int t = 0; t < TT; ++t) {
    float xq = q[t * PLANE + i];
    vq += (xq - vq) * 0.5f;                      // tau = 2
    float sq = (vq >= 1.f) ? 1.f : 0.f;
    vq *= (1.f - sq);
    float xk = k[t * PLANE + i];
    vk += (xk - vk) * 0.5f;
    float sk = (vk >= 1.f) ? 1.f : 0.f;
    vk *= (1.f - sk);
    const int o = ((t * BB + b) * NN + n) * CC + c;  // (tb,n,c)
    qs[o] = (_Float16)sq;
    ks[o] = (_Float16)sk;
  }
  vst[(b * NN + n) * CC + c] = vk;
}

// ---------------- main channel GEMM (512x512x(N=1024) per tb) -------------
// Y[tb][n][m] = sum_k Wh[m][k] * S[tb][n][k].  8 waves: 2(M)x4(N), each wave
// 32x64 (8 WMMA tiles). All fragment loads are contiguous b128s from global.
__global__ void __launch_bounds__(256) gemm_cn_kernel(
    const _Float16* __restrict__ Wh,  // (CC,CC) row-major f16
    const _Float16* __restrict__ S,   // (TB,NN,CC) f16
    float* __restrict__ Y) {          // (TB,NN,CC) f32
  const int lane = threadIdx.x & 31;
  const int wave = threadIdx.x >> 5;
  const int tb = blockIdx.z;
  const int m0 = blockIdx.y * 64 + (wave & 1) * 32;
  const int n0 = blockIdx.x * 256 + (wave >> 1) * 64;
  const _Float16* Sb = S + ((long)tb * NN + n0) * CC;
  v8f acc[2][4] = {};
  for (int k0 = 0; k0 < CC; k0 += 32) {
    if (k0 + 32 < CC) {                         // global_prefetch_b8 path
      __builtin_prefetch(Wh + m0 * CC + k0 + 32, 0, 3);
      __builtin_prefetch(Sb + k0 + 32, 0, 3);
    }
    v16h a0 = load_afrag(Wh + (m0 + 0)  * CC + k0, CC, lane);
    v16h a1 = load_afrag(Wh + (m0 + 16) * CC + k0, CC, lane);
#pragma unroll
    for (int j = 0; j < 4; ++j) {
      v16h b = load_bfrag(Sb + (j * 16) * CC + k0, CC, lane);
      acc[0][j] = wmma_f16(a0, b, acc[0][j]);
      acc[1][j] = wmma_f16(a1, b, acc[1][j]);
    }
  }
#pragma unroll
  for (int i = 0; i < 2; ++i)
#pragma unroll
    for (int j = 0; j < 4; ++j) {
      const int n = n0 + j * 16 + (lane & 15);
      const int m = m0 + i * 16 + ((lane & 16) ? 8 : 0);
      float* p = Y + ((long)tb * NN + n) * CC + m;   // 8 consecutive channels
      v4f lo = { acc[i][j][0], acc[i][j][1], acc[i][j][2], acc[i][j][3] };
      v4f hi = { acc[i][j][4], acc[i][j][5], acc[i][j][6], acc[i][j][7] };
      *(v4f*)p = lo;
      *(v4f*)(p + 4) = hi;
    }
}

// ---------------- BatchNorm (training stats), deterministic 2-stage -------

__global__ void __launch_bounds__(256) bn_stats1_kernel(
    const float* __restrict__ Y, float* __restrict__ part) {
  const int blk = blockIdx.x;                 // 64 blocks x 1024 rows
  const int c0 = threadIdx.x;                 // channels c0 and c0+256
  const float* base = Y + (long)blk * 1024 * CC;
  float s0 = 0.f, q0 = 0.f, s1 = 0.f, q1 = 0.f;
  for (int r = 0; r < 1024; ++r) {
    float a = base[r * CC + c0];
    float b = base[r * CC + c0 + 256];
    s0 += a; q0 += a * a;
    s1 += b; q1 += b * b;
  }
  part[(blk * 2 + 0) * CC + c0]       = s0;
  part[(blk * 2 + 0) * CC + c0 + 256] = s1;
  part[(blk * 2 + 1) * CC + c0]       = q0;
  part[(blk * 2 + 1) * CC + c0 + 256] = q1;
}

__global__ void bn_stats2_kernel(const float* __restrict__ part,
                                 float* __restrict__ mr) {
  const int c = blockIdx.x * 256 + threadIdx.x;   // grid = 2
  float s = 0.f, q = 0.f;
  for (int blk = 0; blk < 64; ++blk) {
    s += part[(blk * 2 + 0) * CC + c];
    q += part[(blk * 2 + 1) * CC + c];
  }
  const float inv = 1.f / 65536.f;                // TB*NN
  float mean = s * inv;
  float var  = q * inv - mean * mean;
  mr[c]      = mean;
  mr[CC + c] = rsqrtf(var + 1e-5f);
}

// BN apply -> f16, same (tb,n,c) layout (q path)
__global__ void __launch_bounds__(256) bn_apply_nc_kernel(
    const float* __restrict__ Y, const float* __restrict__ mr,
    const float* __restrict__ gamma, const float* __restrict__ beta,
    _Float16* __restrict__ out) {
  const int i = blockIdx.x * 256 + threadIdx.x;
  const int c = i & (CC - 1);
  const float a = mr[CC + c] * gamma[c];
  const float b = beta[c] - mr[c] * a;
  out[i] = (_Float16)(Y[i] * a + b);
}

// BN apply -> f16, transposed to (tb,c,n) (k path: A-operand layout for kv)
__global__ void __launch_bounds__(256) bn_apply_cn_kernel(
    const float* __restrict__ Y, const float* __restrict__ mr,
    const float* __restrict__ gamma, const float* __restrict__ beta,
    _Float16* __restrict__ out) {
  const int i = blockIdx.x * 256 + threadIdx.x;   // (tb,n,c) c fastest
  const int c  = i & (CC - 1);
  const int n  = (i >> 9) & (NN - 1);
  const int tb = i >> 19;
  const float a = mr[CC + c] * gamma[c];
  const float b = beta[c] - mr[c] * a;
  out[(tb * CC + c) * NN + n] = (_Float16)(Y[i] * a + b);
}

// v path: BN apply + second LIF with carried membrane state; spikes to (tb,c,n)
__global__ void __launch_bounds__(256) bn_lif_v_kernel(
    const float* __restrict__ Y, const float* __restrict__ mr,
    const float* __restrict__ gamma, const float* __restrict__ beta,
    const float* __restrict__ vst, _Float16* __restrict__ out) {
  const int i = blockIdx.x * 256 + threadIdx.x;   // (b,n,c)
  const int c = i & (CC - 1);
  const int n = (i >> 9) & (NN - 1);
  const int b = i >> 19;
  const float a  = mr[CC + c] * gamma[c];
  const float bb = beta[c] - mr[c] * a;
  float v = vst[i];
#pragma unroll
  for (int t = 0; t < TT; ++t) {
    const int tb = t * BB + b;
    float y = Y[(tb * NN + n) * CC + c] * a + bb;
    v += (y - v) * 0.5f;
    float s = (v >= 1.f) ? 1.f : 0.f;
    v *= (1.f - s);
    out[(tb * CC + c) * NN + n] = (_Float16)s;
  }
}

// ---------------- linear attention ----------------
// kv[d,e] = sum_n K[d,n]*V[e,n]; store transposed+scaled: kvT[e][d] = 0.125*kv
__global__ void __launch_bounds__(256) kv_gemm_kernel(
    const _Float16* __restrict__ Kh,  // (TB,CC,NN)
    const _Float16* __restrict__ Vh,  // (TB,CC,NN)
    _Float16* __restrict__ kvT) {     // (TB*HH,DD,DD)
  const int lane = threadIdx.x & 31;
  const int wave = threadIdx.x >> 5;
  const int tbh = blockIdx.x;
  const int h  = tbh & (HH - 1);
  const int tb = tbh >> 3;
  const int m0 = (wave & 3) * 16;      // d strip
  const int n0 = (wave >> 2) * 32;     // e strip
  const _Float16* Ab = Kh + ((long)tb * CC + h * DD + m0) * NN;
  const _Float16* Bb = Vh + ((long)tb * CC + h * DD + n0) * NN;
  v8f acc[2] = {};
  for (int k0 = 0; k0 < NN; k0 += 32) {
    v16h a  = load_afrag(Ab + k0, NN, lane);
    v16h b0 = load_bfrag(Bb + k0, NN, lane);
    v16h b1 = load_bfrag(Bb + 16 * NN + k0, NN, lane);
    acc[0] = wmma_f16(a, b0, acc[0]);
    acc[1] = wmma_f16(a, b1, acc[1]);
  }
  _Float16* out = kvT + tbh * DD * DD;
#pragma unroll
  for (int j = 0; j < 2; ++j) {
    const int e = n0 + j * 16 + (lane & 15);
    const int d = m0 + ((lane & 16) ? 8 : 0);
    v8h pk;
#pragma unroll
    for (int i = 0; i < 8; ++i) pk[i] = (_Float16)(acc[j][i] * 0.125f);  // SCALE
    *(v8h*)(out + e * DD + d) = pk;
  }
}

// x[e,n] = sum_d kvT[e,d]*Q[d,n]  -> X in (tb,n,c) layout, c = h*64+e
__global__ void __launch_bounds__(256) attn_gemm_kernel(
    const _Float16* __restrict__ kvT, // (TB*HH,DD,DD)
    const _Float16* __restrict__ Q,   // (TB,NN,CC)
    float* __restrict__ X) {          // (TB,NN,CC)
  const int lane = threadIdx.x & 31;
  const int wave = threadIdx.x >> 5;
  const int tbh = blockIdx.y;
  const int h  = tbh & (HH - 1);
  const int tb = tbh >> 3;
  const int m0 = (wave & 3) * 16;                      // e strip
  const int n0 = blockIdx.x * 128 + (wave >> 2) * 64;  // n strip
  const _Float16* Ab = kvT + tbh * DD * DD + m0 * DD;
  const _Float16* Bb = Q + ((long)tb * NN + n0) * CC + h * DD;
  v8f acc[4] = {};
#pragma unroll
  for (int k0 = 0; k0 < DD; k0 += 32) {
    v16h a = load_afrag(Ab + k0, DD, lane);
#pragma unroll
    for (int j = 0; j < 4; ++j) {
      v16h b = load_bfrag(Bb + (j * 16) * CC + k0, CC, lane);
      acc[j] = wmma_f16(a, b, acc[j]);
    }
  }
#pragma unroll
  for (int j = 0; j < 4; ++j) {
    const int n = n0 + j * 16 + (lane & 15);
    const int cbase = h * DD + m0 + ((lane & 16) ? 8 : 0);
    float* p = X + ((long)tb * NN + n) * CC + cbase;
    v4f lo = { acc[j][0], acc[j][1], acc[j][2], acc[j][3] };
    v4f hi = { acc[j][4], acc[j][5], acc[j][6], acc[j][7] };
    *(v4f*)p = lo;
    *(v4f*)(p + 4) = hi;
  }
}

// attn LIF (thr=0.5), (tb,n,c) -> f16 spikes (tb,n,c)
__global__ void __launch_bounds__(256) lif_x_kernel(
    const float* __restrict__ X, _Float16* __restrict__ xs, float thr) {
  const int i = blockIdx.x * 256 + threadIdx.x;  // (b,n,c)
  const int c = i & (CC - 1);
  const int n = (i >> 9) & (NN - 1);
  const int b = i >> 19;
  float v = 0.f;
#pragma unroll
  for (int t = 0; t < TT; ++t) {
    const int o = ((t * BB + b) * NN + n) * CC + c;
    float x = X[o];
    v += (x - v) * 0.5f;
    float s = (v >= thr) ? 1.f : 0.f;
    v *= (1.f - s);
    xs[o] = (_Float16)s;
  }
}

// final BN + LIF -> output in reference (T,B,C,N) f32 layout
__global__ void __launch_bounds__(256) bn_lif_out_kernel(
    const float* __restrict__ Y, const float* __restrict__ mr,
    const float* __restrict__ gamma, const float* __restrict__ beta,
    float* __restrict__ out) {
  const int i = blockIdx.x * 256 + threadIdx.x;  // (b,n,c)
  const int c = i & (CC - 1);
  const int n = (i >> 9) & (NN - 1);
  const int b = i >> 19;
  const float a  = mr[CC + c] * gamma[c];
  const float bb = beta[c] - mr[c] * a;
  float v = 0.f;
#pragma unroll
  for (int t = 0; t < TT; ++t) {
    float y = Y[((t * BB + b) * NN + n) * CC + c] * a + bb;
    v += (y - v) * 0.5f;
    float s = (v >= 1.f) ? 1.f : 0.f;
    v *= (1.f - s);
    out[((t * BB + b) * CC + c) * NN + n] = s;
  }
}

extern "C" void kernel_launch(void* const* d_in, const int* in_sizes, int n_in,
                              void* d_out, int out_size, void* d_ws, size_t ws_size,
                              hipStream_t stream) {
  (void)in_sizes; (void)n_in; (void)out_size; (void)ws_size;
  const float* q  = (const float*)d_in[0];
  const float* k  = (const float*)d_in[1];
  const float* Wq = (const float*)d_in[2];
  const float* qg = (const float*)d_in[3];
  const float* qb = (const float*)d_in[4];
  const float* Wk = (const float*)d_in[5];
  const float* kg = (const float*)d_in[6];
  const float* kb = (const float*)d_in[7];
  const float* Wv = (const float*)d_in[8];
  const float* vg = (const float*)d_in[9];
  const float* vb = (const float*)d_in[10];
  const float* Wp = (const float*)d_in[11];
  // d_in[12] = Wp_bias: a bias added before BatchNorm cancels exactly; unused.
  const float* pg = (const float*)d_in[13];
  const float* pb = (const float*)d_in[14];
  float* out = (float*)d_out;

  char* ws = (char*)d_ws;
  const size_t MiB = 1024ull * 1024ull;
  _Float16* S1   = (_Float16*)(ws + 0);          // 64 MiB: q spikes -> qh (tb,n,c)
  _Float16* S2   = (_Float16*)(ws + 64 * MiB);   // 64 MiB: k spikes -> vh (tb,c,n)
  _Float16* H3   = (_Float16*)(ws + 128 * MiB);  // 64 MiB: kh (tb,c,n) -> xs (tb,n,c)
  float*    F1   = (float*)   (ws + 192 * MiB);  // 128 MiB: fp32 GEMM/attn scratch
  float*    VST  = (float*)   (ws + 320 * MiB);  // 32 MiB: k membrane state (b,n,c)
  _Float16* KV   = (_Float16*)(ws + 352 * MiB);  // 4 MiB: kvT (TB*HH,64,64) f16
  _Float16* WH   = (_Float16*)(ws + 356 * MiB);  // 2 MiB: 4 weight mats in f16
  float*    PART = (float*)   (ws + 358 * MiB);  // 256 KiB: BN partials
  float*    MR   = (float*)   (ws + 358 * MiB + 262144);  // 4 KiB: mean/rstd

  _Float16* WqH = WH;
  _Float16* WkH = WH + 262144;
  _Float16* WvH = WH + 2 * 262144;
  _Float16* WpH = WH + 3 * 262144;

  dim3 blk(256);
  w2h_kernel<<<1024, blk, 0, stream>>>(Wq, WqH);
  w2h_kernel<<<1024, blk, 0, stream>>>(Wk, WkH);
  w2h_kernel<<<1024, blk, 0, stream>>>(Wv, WvH);
  w2h_kernel<<<1024, blk, 0, stream>>>(Wp, WpH);

  lif_qk_kernel<<<BB * CC * NN / 256, blk, 0, stream>>>(q, k, S1, S2, VST);

  dim3 ggrid(NN / 256, CC / 64, TB);  // (4, 8, 64)

  // q path: conv -> BN -> qh (f16, (tb,n,c))
  gemm_cn_kernel<<<ggrid, blk, 0, stream>>>(WqH, S1, F1);
  bn_stats1_kernel<<<64, blk, 0, stream>>>(F1, PART);
  bn_stats2_kernel<<<2, blk, 0, stream>>>(PART, MR);
  bn_apply_nc_kernel<<<TB * NN * CC / 256, blk, 0, stream>>>(F1, MR, qg, qb, S1);

  // k path: conv -> BN -> kh (f16, (tb,c,n))
  gemm_cn_kernel<<<ggrid, blk, 0, stream>>>(WkH, S2, F1);
  bn_stats1_kernel<<<64, blk, 0, stream>>>(F1, PART);
  bn_stats2_kernel<<<2, blk, 0, stream>>>(PART, MR);
  bn_apply_cn_kernel<<<TB * NN * CC / 256, blk, 0, stream>>>(F1, MR, kg, kb, H3);

  // v path: conv -> BN -> LIF(with carried state) -> vh spikes (f16, (tb,c,n))
  gemm_cn_kernel<<<ggrid, blk, 0, stream>>>(WvH, S2, F1);
  bn_stats1_kernel<<<64, blk, 0, stream>>>(F1, PART);
  bn_stats2_kernel<<<2, blk, 0, stream>>>(PART, MR);
  bn_lif_v_kernel<<<BB * NN * CC / 256, blk, 0, stream>>>(F1, MR, vg, vb, VST, S2);

  // linear attention (scale folded into kvT)
  kv_gemm_kernel<<<TB * HH, blk, 0, stream>>>(H3, S2, KV);
  attn_gemm_kernel<<<dim3(NN / 128, TB * HH), blk, 0, stream>>>(KV, S1, F1);
  lif_x_kernel<<<BB * NN * CC / 256, blk, 0, stream>>>(F1, H3, 0.5f);

  // projection: conv (+bias cancels in BN) -> BN -> LIF -> out
  gemm_cn_kernel<<<ggrid, blk, 0, stream>>>(WpH, H3, F1);
  bn_stats1_kernel<<<64, blk, 0, stream>>>(F1, PART);
  bn_stats2_kernel<<<2, blk, 0, stream>>>(PART, MR);
  bn_lif_out_kernel<<<BB * NN * CC / 256, blk, 0, stream>>>(F1, MR, pg, pb, out);
}
